// ViTBlock_49649821942416
// MI455X (gfx1250) — matrix operand order repaired and checked by reading, not compiled, not run on hardware
//
#include <hip/hip_runtime.h>
#include <hip/hip_bf16.h>
#include <math.h>

// ---------------------------------------------------------------------------
// ViT block on MI455X (gfx1250): wave32 + WMMA bf16 + TDM (tensor_load_to_lds)
// double-buffered LDS GEMM. Shapes fixed: B=32, T=197, D=768, H=12, MLP=3072.
// ---------------------------------------------------------------------------

#define BB   32
#define TT   197
#define DD   768
#define HH   12
#define MLPD 3072
#define MROWS (BB * TT)      // 6304
#define THM  (TT * HH)       // 2364  (linear_map K dim)
#define THMP 2368            // padded Wmap / aoT row length (multiple of 8)
#define SP   200             // padded ld for attn / vT rows (multiple of 8)

typedef __attribute__((ext_vector_type(16))) __bf16 v16bf;
typedef __attribute__((ext_vector_type(8)))  float  v8f;
typedef __attribute__((ext_vector_type(4)))  unsigned int u32x4;

// TDM descriptor vector types (probe-verified builtin signatures)
typedef __attribute__((ext_vector_type(4))) unsigned int tdm_u32x4;
typedef __attribute__((ext_vector_type(8))) int          tdm_i32x8;
typedef __attribute__((ext_vector_type(4))) int          tdm_i32x4;

#if __has_builtin(__builtin_amdgcn_tensor_load_to_lds)
#define HAVE_TDM 1
#else
#define HAVE_TDM 0
#endif

// GEMM tiling: block tile 64(M) x 128(N), K step 32. 256 threads = 8 wave32.
// Wave w: wm = w&1 (two 32-row strips), wn = w>>1 (four 32-col strips).
// Each wave: 32x32 output = 2 A-frags x 2 B-frags -> 4 WMMA per K step.
#define BM 64
#define BN 128
#define BK 32
#define LDS_ST (BK + 8)      // 40 bf16 = 80 B row stride (16B aligned)

enum { M_QK = 0, M_V = 1, M_SCORES = 2, M_AO = 3, M_MAP = 4, M_MLP1 = 5, M_MLP2 = 6 };

union Frag  { v16bf v; u32x4 q[2]; };
union Pack8 { u32x4 q; __bf16 h[8]; };

#if HAVE_TDM
// ---------------------------------------------------------------------------
// Issue one TDM 2D tile load: global (rows x cols, row stride `stride` elems,
// 2-byte elems) -> LDS at byte offset lds_addr, padding +4 DWORDs after every
// 16 DWORDs stored (=> LDS row stride 40 bf16). tensor_dim0/1 are the valid
// extents from the tile origin: TDM zero-fills reads beyond them (replaces all
// K/M/N edge guards). D# packing per CDNA5 ISA ch.8 (group0 / group1).
// ---------------------------------------------------------------------------
__device__ __forceinline__ void tdm_load_2d(unsigned lds_addr, const void* gptr,
                                            unsigned tdim0, unsigned tdim1,
                                            unsigned tile0, unsigned tile1,
                                            unsigned stride_elems)
{
    const unsigned long long ga = (unsigned long long)gptr;
    tdm_u32x4 g0;
    g0.x = 1u;                                   // count=1, user descriptor
    g0.y = lds_addr;                             // lds_addr[31:0]
    g0.z = (unsigned)ga;                         // global_addr[31:0]
    g0.w = (unsigned)((ga >> 32) & 0x01FFFFFFu)  // global_addr[56:32]
         | 0x80000000u;                          // type=2 ("image")

    tdm_i32x8 g1;
    // data_size=1 (2B) | pad_enable | pad_interval=3 (16 DW) | pad_amount=3 (4 DW)
    g1[0] = (int)((1u << 16) | (1u << 20) | (3u << 22) | (3u << 25));
    g1[1] = (int)((tdim0 & 0xFFFFu) << 16);                    // tensor_dim0[15:0]
    g1[2] = (int)((tdim0 >> 16) | ((tdim1 & 0xFFFFu) << 16));  // dim0[31:16] | dim1[15:0]
    g1[3] = (int)((tdim1 >> 16) | (tile0 << 16));              // dim1[31:16] | tile_dim0
    g1[4] = (int)(tile1 & 0xFFFFu);                            // tile_dim1 (tile_dim2=0)
    g1[5] = (int)stride_elems;                                 // tensor_dim0_stride[31:0]
    g1[6] = 0;                                                 // stride[47:32] | dim1_stride
    g1[7] = 0;

    const tdm_i32x4 z4 = {0, 0, 0, 0};
#if defined(__clang_major__) && __clang_major__ >= 23
    const tdm_i32x8 z8 = {0, 0, 0, 0, 0, 0, 0, 0};
    __builtin_amdgcn_tensor_load_to_lds(g0, g1, z4, z4, z8, 0);
#else
    __builtin_amdgcn_tensor_load_to_lds(g0, g1, z4, z4, 0);
#endif
}
#else
// Fallback: register-staged fetch (clamped vector loads; branchless K tail).
__device__ __forceinline__ void fetch_tile(
    const __bf16* __restrict__ A, int lda,
    const __bf16* __restrict__ Bw, int ldb,
    int M, int N, int K, int m0, int n0, int k0,
    int ar, int ac, int br, int bc,
    u32x4& sa, u32x4& sb0, u32x4& sb1)
{
    int gm = m0 + ar; if (gm >= M) gm = M - 1;
    int gn = n0 + br; if (gn >= N) gn = N - 1;
    if (k0 + BK <= K) {
        sa  = *(const u32x4*)(A  + (size_t)gm * lda + k0 + ac);
        const __bf16* src = Bw + (size_t)gn * ldb + k0 + bc;
        sb0 = *(const u32x4*)src;
        sb1 = *(const u32x4*)(src + 8);
    } else {
        Pack8 pa, pb0, pb1;
        #pragma unroll
        for (int j = 0; j < 8; ++j) {
            const int ka = k0 + ac + j;
            const __bf16 va = A[(size_t)gm * lda + (ka < K ? ka : K - 1)];
            pa.h[j] = (ka < K) ? va : (__bf16)0.0f;
            const int kb = k0 + bc + j;
            const __bf16 vb = Bw[(size_t)gn * ldb + (kb < K ? kb : K - 1)];
            pb0.h[j] = (kb < K) ? vb : (__bf16)0.0f;
            const int kb2 = kb + 8;
            const __bf16 vb2 = Bw[(size_t)gn * ldb + (kb2 < K ? kb2 : K - 1)];
            pb1.h[j] = (kb2 < K) ? vb2 : (__bf16)0.0f;
        }
        sa = pa.q; sb0 = pb0.q; sb1 = pb1.q;
    }
}
#endif

template <int MODE>
__global__ __launch_bounds__(256)
void gemm_k(const __bf16* __restrict__ Abase, int lda,
            const __bf16* __restrict__ Bbase, int ldb,
            void* __restrict__ Cbase,
            const float* __restrict__ bias,
            const float* __restrict__ extra,
            int M, int N, int K)
{
    __shared__ __attribute__((aligned(16))) __bf16 As[2][BM * LDS_ST];
    __shared__ __attribute__((aligned(16))) __bf16 Bs[2][BN * LDS_ST];

    const int tid = threadIdx.x;
    const int z   = blockIdx.z;
    const int m0  = blockIdx.x * BM;
    const int n0  = blockIdx.y * BN;

    const __bf16* A  = Abase;
    const __bf16* Bw = Bbase;
    if constexpr (MODE == M_QK || MODE == M_V)  { Bw += (size_t)z * DD * DD; }
    if constexpr (MODE == M_SCORES)             { A  += (size_t)z * TT * DD;
                                                  Bw += (size_t)z * TT * DD; }
    if constexpr (MODE == M_AO)                 { A  += (size_t)z * TT * SP;
                                                  Bw += (size_t)z * DD * SP; }
    if constexpr (MODE == M_MAP)                { Bw += (size_t)z * DD * THMP; }

    const int wave = tid >> 5;
    const int lane = tid & 31;
    const int wm   = wave & 1;
    const int wn   = wave >> 1;          // 0..3
    const int half = lane >> 4;
    const int r    = lane & 15;

    v8f acc[2][2] = {{v8f{}, v8f{}}, {v8f{}, v8f{}}};

    const int ktiles = (K + BK - 1) / BK;

#if HAVE_TDM
    // TDM double-buffer: wave 0 drives the Tensor Data Mover; compute waves
    // only touch LDS. Two barriers per K step; tile k+1 DMA overlaps WMMA of k.
    if (wave == 0) {
        tdm_load_2d((unsigned)(size_t)&As[0][0], A + (size_t)m0 * lda,
                    (unsigned)K, (unsigned)(M - m0), BK, BM, (unsigned)lda);
        tdm_load_2d((unsigned)(size_t)&Bs[0][0], Bw + (size_t)n0 * ldb,
                    (unsigned)K, (unsigned)(N - n0), BK, BN, (unsigned)ldb);
    }
    for (int kt = 0; kt < ktiles; ++kt) {
        __syncthreads();                 // all waves done reading buf[nxt]
        const int cur = kt & 1;
        if (wave == 0) {
            if (kt + 1 < ktiles) {
                const int nxt = cur ^ 1;
                const int k1 = (kt + 1) * BK;
                tdm_load_2d((unsigned)(size_t)&As[nxt][0],
                            A + (size_t)m0 * lda + k1,
                            (unsigned)(K - k1), (unsigned)(M - m0), BK, BM,
                            (unsigned)lda);
                tdm_load_2d((unsigned)(size_t)&Bs[nxt][0],
                            Bw + (size_t)n0 * ldb + k1,
                            (unsigned)(K - k1), (unsigned)(N - n0), BK, BN,
                            (unsigned)ldb);
                __builtin_amdgcn_s_wait_tensorcnt(2);   // tile kt landed
            } else {
                __builtin_amdgcn_s_wait_tensorcnt(0);
            }
        }
        __syncthreads();                 // tile kt visible to all waves
#else
    // Fallback: per-thread register staging, one barrier per K step.
    const int ar = tid >> 2, ac = (tid & 3) * 8;
    const int br = tid >> 1, bc = (tid & 1) * 16;
    u32x4 sa, sb0, sb1;
    fetch_tile(A, lda, Bw, ldb, M, N, K, m0, n0, 0, ar, ac, br, bc, sa, sb0, sb1);
    {
        __bf16* da = &As[0][ar * LDS_ST + ac];
        *(u32x4*)da = sa;
        __bf16* db = &Bs[0][br * LDS_ST + bc];
        *(u32x4*)db = sb0; *(u32x4*)(db + 8) = sb1;
    }
    for (int kt = 0; kt < ktiles; ++kt) {
        __syncthreads();
        const int cur = kt & 1;
        const bool more = (kt + 1) < ktiles;
        if (more)
            fetch_tile(A, lda, Bw, ldb, M, N, K, m0, n0, (kt + 1) * BK,
                       ar, ac, br, bc, sa, sb0, sb1);
#endif

        // ---- fragments (contiguous k order, k = 16*half + s, same in A and B) ----
        // A 16x32: lanes 0-15 hold K 0-15 of row r; lanes 16-31 hold K 16-31.
        Frag af0, af1;
        {
            const __bf16* a0 = &As[cur][(wm * 32 + r) * LDS_ST];
            af0.q[0] = *(const u32x4*)(a0 + half * 16);
            af0.q[1] = *(const u32x4*)(a0 + 8 + half * 16);
            const __bf16* a1 = a0 + 16 * LDS_ST;
            af1.q[0] = *(const u32x4*)(a1 + half * 16);
            af1.q[1] = *(const u32x4*)(a1 + 8 + half * 16);
        }
        // B 32x16: lanes 0-15 hold K 0-15 of col N=r; lanes 16-31 hold K 16-31.
        Frag bf0, bf1;
        {
            const __bf16* b0 = &Bs[cur][(wn * 32 + r) * LDS_ST + half * 16];
            bf0.q[0] = *(const u32x4*)(b0);
            bf0.q[1] = *(const u32x4*)(b0 + 8);
            const __bf16* b1 = &Bs[cur][(wn * 32 + 16 + r) * LDS_ST + half * 16];
            bf1.q[0] = *(const u32x4*)(b1);
            bf1.q[1] = *(const u32x4*)(b1 + 8);
        }

        acc[0][0] = __builtin_amdgcn_wmma_f32_16x16x32_bf16(
            false, af0.v, false, bf0.v, (short)0, acc[0][0], false, false);
        acc[0][1] = __builtin_amdgcn_wmma_f32_16x16x32_bf16(
            false, af0.v, false, bf1.v, (short)0, acc[0][1], false, false);
        acc[1][0] = __builtin_amdgcn_wmma_f32_16x16x32_bf16(
            false, af1.v, false, bf0.v, (short)0, acc[1][0], false, false);
        acc[1][1] = __builtin_amdgcn_wmma_f32_16x16x32_bf16(
            false, af1.v, false, bf1.v, (short)0, acc[1][1], false, false);

#if !HAVE_TDM
        if (more) {
            const int nxt = cur ^ 1;
            __bf16* da = &As[nxt][ar * LDS_ST + ac];
            *(u32x4*)da = sa;
            __bf16* db = &Bs[nxt][br * LDS_ST + bc];
            *(u32x4*)db = sb0; *(u32x4*)(db + 8) = sb1;
        }
#endif
    }

    // ---- epilogue (C layout: VGPR g, lanes 0-15 -> M=g, N=lane; lanes 16-31 -> M=8+g) ----
    #pragma unroll
    for (int ms = 0; ms < 2; ++ms) {
        #pragma unroll
        for (int ns = 0; ns < 2; ++ns) {
            #pragma unroll
            for (int g = 0; g < 8; ++g) {
                const int gm = m0 + wm * 32 + ms * 16 + half * 8 + g;
                const int gn = n0 + wn * 32 + ns * 16 + r;
                if (gm >= M || gn >= N) continue;
                const float v = acc[ms][ns][g];

                if constexpr (MODE == M_QK) {
                    const int b = gm / TT, t = gm - b * TT;
                    const float vb = v + bias[(size_t)z * DD + gn];
                    ((__bf16*)Cbase)[((size_t)(b * HH + z) * TT + t) * DD + gn] = (__bf16)vb;
                } else if constexpr (MODE == M_V) {
                    const int b = gm / TT, t = gm - b * TT;
                    const float vb = v + bias[(size_t)z * DD + gn];
                    ((__bf16*)Cbase)[((size_t)(b * HH + z) * DD + gn) * SP + t] = (__bf16)vb;
                } else if constexpr (MODE == M_SCORES) {
                    ((float*)Cbase)[(size_t)z * TT * TT + (size_t)gm * TT + gn] = v;
                } else if constexpr (MODE == M_AO) {
                    const int b = z / HH, h = z - b * HH;
                    ((__bf16*)Cbase)[((size_t)b * DD + gn) * THMP + h * TT + gm] = (__bf16)v;
                } else if constexpr (MODE == M_MAP) {
                    const size_t o = ((size_t)z * TT + gm) * DD + gn;
                    ((float*)Cbase)[o] = v + bias[gm] + extra[o];      // + bmap[t] + x
                } else if constexpr (MODE == M_MLP1) {
                    const float u = v + bias[gn];
                    const float ge = 0.5f * u * (1.0f + erff(u * 0.7071067811865476f));
                    ((__bf16*)Cbase)[(size_t)gm * MLPD + gn] = (__bf16)ge;
                } else { // M_MLP2
                    const size_t o = (size_t)gm * DD + gn;
                    ((float*)Cbase)[o] = v + bias[gn] + extra[o];      // + b2 + h2
                }
            }
        }
    }
}

// ---------------------------------------------------------------------------
// fp32 -> bf16 conversion (weights)
// ---------------------------------------------------------------------------
__global__ __launch_bounds__(256)
void cvt_bf16_k(const float* __restrict__ in, __bf16* __restrict__ out, long long n)
{
    long long i = (long long)blockIdx.x * blockDim.x + threadIdx.x;
    const long long stride = (long long)gridDim.x * blockDim.x;
    for (; i < n; i += stride) out[i] = (__bf16)in[i];
}

// fp32 -> bf16 with row padding (zero fill) : Wmap 2364 -> 2368 cols
__global__ __launch_bounds__(256)
void cvt_pad_k(const float* __restrict__ in, __bf16* __restrict__ out,
               int rows, int incols, int outcols)
{
    long long n = (long long)rows * outcols;
    long long i = (long long)blockIdx.x * blockDim.x + threadIdx.x;
    const long long stride = (long long)gridDim.x * blockDim.x;
    for (; i < n; i += stride) {
        const int rr = (int)(i / outcols);
        const int cc = (int)(i - (long long)rr * outcols);
        out[i] = (cc < incols) ? (__bf16)in[(size_t)rr * incols + cc] : (__bf16)0.0f;
    }
}

// ---------------------------------------------------------------------------
// LayerNorm: one 256-thread block per row of D=768. Emits bf16 (and optional f32).
// ---------------------------------------------------------------------------
__global__ __launch_bounds__(256)
void layernorm_k(const float* __restrict__ x,
                 const float* __restrict__ gamma, const float* __restrict__ beta,
                 __bf16* __restrict__ obf, float* __restrict__ of)
{
    const int row = blockIdx.x;
    const float* xr = x + (size_t)row * DD;
    const int t = threadIdx.x;
    float v0 = xr[t], v1 = xr[t + 256], v2 = xr[t + 512];
    float s  = v0 + v1 + v2;
    float ss = v0 * v0 + v1 * v1 + v2 * v2;
    #pragma unroll
    for (int off = 16; off; off >>= 1) {
        s  += __shfl_xor(s,  off, 32);
        ss += __shfl_xor(ss, off, 32);
    }
    __shared__ float ps[8], pss[8];
    const int wave = t >> 5, lane = t & 31;
    if (lane == 0) { ps[wave] = s; pss[wave] = ss; }
    __syncthreads();
    if (t == 0) {
        float a = 0.f, c = 0.f;
        #pragma unroll
        for (int i = 0; i < 8; ++i) { a += ps[i]; c += pss[i]; }
        ps[0] = a; pss[0] = c;
    }
    __syncthreads();
    const float mean = ps[0] * (1.0f / DD);
    const float var  = pss[0] * (1.0f / DD) - mean * mean;
    const float rstd = rsqrtf(var + 1e-5f);
    #pragma unroll
    for (int j = 0; j < 3; ++j) {
        const int d = t + j * 256;
        const float xv = (j == 0) ? v0 : (j == 1) ? v1 : v2;
        const float y = (xv - mean) * rstd * gamma[d] + beta[d];
        obf[(size_t)row * DD + d] = (__bf16)y;
        if (of) of[(size_t)row * DD + d] = y;
    }
}

// ---------------------------------------------------------------------------
// Softmax over score rows (len 197), scale 1/sqrt(196) folded in. 1 wave / row.
// ---------------------------------------------------------------------------
__global__ __launch_bounds__(256)
void softmax_k(const float* __restrict__ scores, __bf16* __restrict__ attn)
{
    const int row = blockIdx.x * 8 + (threadIdx.x >> 5);
    if (row >= BB * HH * TT) return;
    const int lane = threadIdx.x & 31;
    const float* sr = scores + (size_t)row * TT;
    float v[7];
    float mx = -1e30f;
    #pragma unroll
    for (int j = 0; j < 7; ++j) {
        const int c = lane + j * 32;
        v[j] = (c < TT) ? sr[c] : -1e30f;
        mx = fmaxf(mx, v[j]);
    }
    #pragma unroll
    for (int off = 16; off; off >>= 1) mx = fmaxf(mx, __shfl_xor(mx, off, 32));
    const float inv = 1.0f / 14.0f;   // 1/sqrt(T-1) = 1/sqrt(196)
    float sum = 0.f;
    #pragma unroll
    for (int j = 0; j < 7; ++j) { v[j] = expf((v[j] - mx) * inv); sum += v[j]; }
    #pragma unroll
    for (int off = 16; off; off >>= 1) sum += __shfl_xor(sum, off, 32);
    const float rs = 1.0f / sum;
    #pragma unroll
    for (int j = 0; j < 7; ++j) {
        const int c = lane + j * 32;
        if (c < TT) attn[(size_t)row * SP + c] = (__bf16)(v[j] * rs);
    }
}

// ---------------------------------------------------------------------------
// Host-side orchestration
// ---------------------------------------------------------------------------
static inline char* carve(char*& p, size_t bytes)
{
    char* r = p;
    p += (bytes + 255) & ~(size_t)255;
    return r;
}

extern "C" void kernel_launch(void* const* d_in, const int* in_sizes, int n_in,
                              void* d_out, int out_size, void* d_ws, size_t ws_size,
                              hipStream_t stream)
{
    const float* x     = (const float*)d_in[0];
    const float* ln1_g = (const float*)d_in[1];
    const float* ln1_b = (const float*)d_in[2];
    const float* Wq    = (const float*)d_in[3];
    const float* bq    = (const float*)d_in[4];
    const float* Wk    = (const float*)d_in[5];
    const float* bk    = (const float*)d_in[6];
    const float* Wv    = (const float*)d_in[7];
    const float* bv    = (const float*)d_in[8];
    const float* Wmap  = (const float*)d_in[9];
    const float* bmap  = (const float*)d_in[10];
    const float* ln2_g = (const float*)d_in[11];
    const float* ln2_b = (const float*)d_in[12];
    const float* W1    = (const float*)d_in[13];
    const float* b1    = (const float*)d_in[14];
    const float* W2    = (const float*)d_in[15];
    const float* b2    = (const float*)d_in[16];

    // ---- workspace carve ----
    char* p = (char*)d_ws;
    const size_t nWqkv  = (size_t)HH * DD * DD;
    const size_t nWmapP = (size_t)TT * THMP;
    const size_t nW12   = (size_t)MLPD * DD;
    const size_t nQKV   = (size_t)BB * HH * TT * DD;
    const size_t nVT    = (size_t)BB * HH * DD * SP;
    const size_t nSC    = (size_t)BB * HH * TT * TT;
    const size_t nAT    = (size_t)BB * HH * TT * SP;
    const size_t nAO    = (size_t)BB * DD * THMP;
    const size_t nACT   = (size_t)MROWS * DD;
    const size_t nMID   = (size_t)MROWS * MLPD;

    __bf16* hbf    = (__bf16*)carve(p, nACT * 2);
    __bf16* Wq_bf  = (__bf16*)carve(p, nWqkv * 2);
    __bf16* Wk_bf  = (__bf16*)carve(p, nWqkv * 2);
    __bf16* Wv_bf  = (__bf16*)carve(p, nWqkv * 2);
    __bf16* Wm_bf  = (__bf16*)carve(p, nWmapP * 2);
    __bf16* W1_bf  = (__bf16*)carve(p, nW12 * 2);
    __bf16* W2_bf  = (__bf16*)carve(p, nW12 * 2);
    __bf16* qbf    = (__bf16*)carve(p, nQKV * 2);
    __bf16* kbf    = (__bf16*)carve(p, nQKV * 2);
    __bf16* vT     = (__bf16*)carve(p, nVT * 2);
    float*  sc     = (float*) carve(p, nSC * 4);
    __bf16* attn   = (__bf16*)carve(p, nAT * 2);
    __bf16* aoT    = (__bf16*)carve(p, nAO * 2);
    float*  outpre = (float*) carve(p, nACT * 4);
    float*  h2f    = (float*) carve(p, nACT * 4);
    __bf16* h2bf   = (__bf16*)carve(p, nACT * 2);
    __bf16* mid    = (__bf16*)carve(p, nMID * 2);

    if ((size_t)(p - (char*)d_ws) > ws_size) return;   // insufficient scratch

    // ---- 1) weights to bf16 ----
    cvt_bf16_k<<<1024, 256, 0, stream>>>(Wq, Wq_bf, (long long)nWqkv);
    cvt_bf16_k<<<1024, 256, 0, stream>>>(Wk, Wk_bf, (long long)nWqkv);
    cvt_bf16_k<<<1024, 256, 0, stream>>>(Wv, Wv_bf, (long long)nWqkv);
    cvt_pad_k <<<1024, 256, 0, stream>>>(Wmap, Wm_bf, TT, THM, THMP);
    cvt_bf16_k<<<1024, 256, 0, stream>>>(W1, W1_bf, (long long)nW12);
    cvt_bf16_k<<<1024, 256, 0, stream>>>(W2, W2_bf, (long long)nW12);

    // ---- 2) LN1 ----
    layernorm_k<<<MROWS, 256, 0, stream>>>(x, ln1_g, ln1_b, hbf, nullptr);

    // ---- 3) QKV projections (grid.z = head) ----
    const dim3 gQKV((MROWS + BM - 1) / BM, DD / BN, HH);               // (99, 6, 12)
    gemm_k<M_QK><<<gQKV, 256, 0, stream>>>(hbf, DD, Wq_bf, DD, qbf, bq, nullptr, MROWS, DD, DD);
    gemm_k<M_QK><<<gQKV, 256, 0, stream>>>(hbf, DD, Wk_bf, DD, kbf, bk, nullptr, MROWS, DD, DD);
    gemm_k<M_V> <<<gQKV, 256, 0, stream>>>(hbf, DD, Wv_bf, DD, vT,  bv, nullptr, MROWS, DD, DD);

    // ---- 4) scores = q.k^T (scale folded into softmax) ----
    const dim3 gSC((TT + BM - 1) / BM, (TT + BN - 1) / BN, BB * HH);   // (4, 2, 384)
    gemm_k<M_SCORES><<<gSC, 256, 0, stream>>>(qbf, DD, kbf, DD, sc, nullptr, nullptr, TT, TT, DD);

    // ---- 5) softmax -> attn (bf16, padded ld) ----
    softmax_k<<<(BB * HH * TT) / 8, 256, 0, stream>>>(sc, attn);

    // ---- 6) ao = attn.v, stored transposed into aoT [B, D, H*T] ----
    const dim3 gAO((TT + BM - 1) / BM, DD / BN, BB * HH);              // (4, 6, 384)
    gemm_k<M_AO><<<gAO, 256, 0, stream>>>(attn, SP, vT, SP, aoT, nullptr, nullptr, TT, DD, TT);

    // ---- 7) msa = Wmap . result + bmap ; outpre = x + msa ----
    const dim3 gMAP((TT + BM - 1) / BM, DD / BN, BB);                  // (4, 6, 32)
    gemm_k<M_MAP><<<gMAP, 256, 0, stream>>>(Wm_bf, THMP, aoT, THMP, outpre, bmap, x, TT, DD, THM);

    // ---- 8) LN2 (keeps fp32 copy for final residual) ----
    layernorm_k<<<MROWS, 256, 0, stream>>>(outpre, ln2_g, ln2_b, h2bf, h2f);

    // ---- 9) MLP ----
    const dim3 gM1((MROWS + BM - 1) / BM, MLPD / BN, 1);               // (99, 24, 1)
    gemm_k<M_MLP1><<<gM1, 256, 0, stream>>>(h2bf, DD, W1_bf, DD, mid, b1, nullptr, MROWS, MLPD, DD);
    const dim3 gM2((MROWS + BM - 1) / BM, DD / BN, 1);                 // (99, 6, 1)
    gemm_k<M_MLP2><<<gM2, 256, 0, stream>>>(mid, MLPD, W2_bf, MLPD, (float*)d_out, b2, h2f,
                                            MROWS, DD, MLPD);
}
